// GraphNN_73237782332018
// MI455X (gfx1250) — matrix operand order repaired and checked by reading, not compiled, not run on hardware
//
#include <hip/hip_runtime.h>

// ---------------- problem constants (from reference) ----------------
#define N_NODES   100000
#define N_EDGES   1600000
#define FDIM      128          // DIN == HID == 128
#define NGRAPH    100
#define NODES_PG  1000         // N / G, graphs are contiguous equal-size
#define OUTC      10
#define NLAYER    4            // L - 1 GIN layers
#define BN_EPS    1e-5f

// ---------------- WMMA vector types (gfx1250, wave32) ----------------
typedef __attribute__((ext_vector_type(16))) __bf16 v16bf;
typedef __attribute__((ext_vector_type(8)))  __bf16 v8bf;
typedef __attribute__((ext_vector_type(8)))  float  v8f;

struct Ptr8 { const float* p[8]; };
struct Ptr5 { const float* p[5]; };

// ---------------- utility: zero a float buffer ----------------
__global__ void k_zero(float* __restrict__ p, int n) {
    int i = blockIdx.x * 256 + threadIdx.x;
    if (i < n) p[i] = 0.0f;
}

// ---------------- repack 128x128 f32 weights -> bf16 WMMA B-fragments ----
// Layout per matrix: [kt(4)][nt(8)][lane(32)][16 halves]; lane: n=lane&15,
// half=lane>>4; fragment half j pairs: b[2v]=W[kt*32+2v+8*half+8*(v>>2)][nt*16+n].
__global__ void k_repack(__bf16* __restrict__ wp, Ptr8 mats) {
    int mat  = blockIdx.x >> 5;          // 8 matrices
    int tile = blockIdx.x & 31;          // 4 kt x 8 nt
    int kt = tile >> 3, nt = tile & 7;
    int lane = threadIdx.x;              // 32 threads
    const float* W = mats.p[mat];
    int n = lane & 15, half = lane >> 4;
    __bf16* dst = wp + (size_t)mat * 16384
                     + (size_t)((kt * 8 + nt) * 32 + lane) * 16;
#pragma unroll
    for (int v = 0; v < 8; ++v) {
        int k0 = kt * 32 + 2 * v + 8 * half + 8 * (v >> 2);
        dst[2 * v]     = (__bf16)W[(size_t)k0 * FDIM + nt * 16 + n];
        dst[2 * v + 1] = (__bf16)W[(size_t)(k0 + 1) * FDIM + nt * 16 + n];
    }
}

// ---------------- score init: sum of 5 prediction biases ----------------
__global__ void k_init_score(float* __restrict__ score, Ptr5 bs) {
    int i = blockIdx.x * 256 + threadIdx.x;
    if (i < NGRAPH * OUTC) {
        int o = i % OUTC;
        score[i] = bs.p[0][o] + bs.p[1][o] + bs.p[2][o] + bs.p[3][o] + bs.p[4][o];
    }
}

// ---------------- pooled = (1 + eps[l]) * h ----------------
__global__ void k_scale_init(float* __restrict__ pooled, const float* __restrict__ h,
                             const float* __restrict__ eps, int l) {
    int i = blockIdx.x * 256 + threadIdx.x;   // exact grid: N*128/256
    pooled[i] = (1.0f + eps[l]) * h[i];
}

// ---------------- edge scatter: pooled[row] += h[col] ----------------
// One 32-lane wave per edge; each lane moves 4 floats (float4 gather, 4
// global_atomic_add_f32). Feature matrix is L2-resident (51 MB << 192 MB).
__global__ void k_edge_scatter(float* __restrict__ pooled, const float* __restrict__ h,
                               const int* __restrict__ row, const int* __restrict__ col) {
    int gid   = blockIdx.x * 256 + threadIdx.x;  // exact grid: E*32/256
    int e     = gid >> 5;
    int lane4 = (gid & 31) * 4;
    int r = row[e], c = col[e];
    const float4 v = *(const float4*)(h + (size_t)c * FDIM + lane4);
    float* dst = pooled + (size_t)r * FDIM + lane4;
    atomicAdd(dst + 0, v.x);
    atomicAdd(dst + 1, v.y);
    atomicAdd(dst + 2, v.z);
    atomicAdd(dst + 3, v.w);
}

// ---------------- f32 -> bf16 staging ----------------
__global__ void k_tobf16(__bf16* __restrict__ dst, const float* __restrict__ src) {
    int i = blockIdx.x * 256 + threadIdx.x;
    dst[i] = (__bf16)src[i];
}

// ---------------- GEMM: out(N x 128) = A(bf16) @ Wp + bias, f32 accum ----
// One wave per 16x16 tile; block of 8 waves covers one 16-row strip.
// A fragment per lane: two contiguous 16B loads per K-step (ISA 16-bit A
// layout); B fragment prepacked to two contiguous 16B loads.
__global__ __launch_bounds__(256)
void k_gemm_bias(float* __restrict__ out, const __bf16* __restrict__ A,
                 const __bf16* __restrict__ Bp, const float* __restrict__ bias) {
    int wave = threadIdx.x >> 5;          // nt = wave (0..7)
    int lane = threadIdx.x & 31;
    int row0 = blockIdx.x * 16;           // N/16 = 6250 blocks, exact
    int l16  = lane & 15;                 // A: row-in-tile; B/D: col-in-tile
    int half = lane >> 4;

    v8f c = {0.f, 0.f, 0.f, 0.f, 0.f, 0.f, 0.f, 0.f};
    const __bf16* arow  = A  + (size_t)(row0 + l16) * FDIM + half * 8;
    const __bf16* bbase = Bp + (size_t)(wave * 32 + lane) * 16;

#pragma unroll
    for (int kt = 0; kt < 4; ++kt) {
        v8bf alo = *(const v8bf*)(arow + kt * 32);
        v8bf ahi = *(const v8bf*)(arow + kt * 32 + 16);
        const __bf16* bt = bbase + (size_t)kt * 4096;   // kt stride = 8*32*16
        v8bf blo = *(const v8bf*)(bt);
        v8bf bhi = *(const v8bf*)(bt + 8);
        v16bf a, b;
#pragma unroll
        for (int i = 0; i < 8; ++i) {
            a[i] = alo[i]; a[8 + i] = ahi[i];
            b[i] = blo[i]; b[8 + i] = bhi[i];
        }
        c = __builtin_amdgcn_wmma_f32_16x16x32_bf16(
                false, a, false, b, (short)0, c, false, false);
    }

    float bv = bias[wave * 16 + l16];
#pragma unroll
    for (int r = 0; r < 8; ++r) {
        out[(size_t)(row0 + r + 8 * half) * FDIM + wave * 16 + l16] = c[r] + bv;
    }
}

// ---------------- per-column sum / sumsq (BN statistics) ----------------
__global__ void k_colstats(float* __restrict__ stats, const float* __restrict__ t) {
    int c  = threadIdx.x & 127;
    int g2 = threadIdx.x >> 7;
    int r0 = blockIdx.x * 100 + g2 * 50;   // 1000 blocks x 100 rows, exact
    float s = 0.f, s2 = 0.f;
    for (int r = r0; r < r0 + 50; ++r) {
        float v = t[(size_t)r * FDIM + c];
        s += v; s2 += v * v;
    }
    atomicAdd(&stats[c], s);
    atomicAdd(&stats[128 + c], s2);
}

__device__ inline float bn_relu(float v, const float* stats,
                                const float* gam, const float* bet, int c) {
    float mu  = stats[c] * (1.0f / N_NODES);
    float var = stats[128 + c] * (1.0f / N_NODES) - mu * mu;   // biased var
    float x = gam[c] * (v - mu) * rsqrtf(var + BN_EPS) + bet[c];
    return x > 0.f ? x : 0.f;
}

__global__ void k_bn_relu_bf16(__bf16* __restrict__ out, const float* __restrict__ t,
                               const float* __restrict__ stats,
                               const float* __restrict__ gam, const float* __restrict__ bet) {
    int i = blockIdx.x * 256 + threadIdx.x;
    out[i] = (__bf16)bn_relu(t[i], stats, gam, bet, i & 127);
}

__global__ void k_bn_relu_f32(float* __restrict__ out, const float* __restrict__ t,
                              const float* __restrict__ stats,
                              const float* __restrict__ gam, const float* __restrict__ bet) {
    int i = blockIdx.x * 256 + threadIdx.x;
    out[i] = bn_relu(t[i], stats, gam, bet, i & 127);
}

// ---------------- graph sum-pool + 128->10 readout ----------------
// One block per graph; 128 lanes sweep each row (coalesced), 2 row groups.
__global__ __launch_bounds__(256)
void k_pool_pred(const float* __restrict__ h, const float* __restrict__ W,
                 float* __restrict__ score) {
    __shared__ float ss[256];
    int g  = blockIdx.x;
    int c  = threadIdx.x & 127;
    int g2 = threadIdx.x >> 7;
    int r0 = g * NODES_PG + g2 * (NODES_PG / 2);
    float s = 0.f;
    for (int r = r0; r < r0 + NODES_PG / 2; ++r)
        s += h[(size_t)r * FDIM + c];
    ss[threadIdx.x] = s;
    __syncthreads();
    if (threadIdx.x < OUTC) {
        int o = threadIdx.x;
        float acc = 0.f;
        for (int cc = 0; cc < FDIM; ++cc)
            acc += (ss[cc] + ss[cc + 128]) * W[cc * OUTC + o];
        score[g * OUTC + o] += acc;   // safe: one block per g, kernels ordered
    }
}

// ---------------- host orchestration ----------------
// Input order (JAX pytree flatten, sorted dict keys inside params):
//  [0]=x, [1]=eps(4), [2+8l..]={W1,W2,b1,b2,bn1_b,bn1_g,bn_b,bn_g} x4,
//  [34+2l]={predW,predb} x5, [44]=edge_index(2E), [45]=graph_ids.
extern "C" void kernel_launch(void* const* d_in, const int* in_sizes, int n_in,
                              void* d_out, int out_size, void* d_ws, size_t ws_size,
                              hipStream_t stream) {
    const float* x   = (const float*)d_in[0];
    const float* eps = (const float*)d_in[1];
    const int* edges = (const int*)d_in[44];
    const int* erow  = edges;
    const int* ecol  = edges + N_EDGES;

    char* w = (char*)d_ws;
    float*  hbuf  = (float*)w;  w += (size_t)N_NODES * FDIM * 4;   // 51.2 MB
    float*  tbuf  = (float*)w;  w += (size_t)N_NODES * FDIM * 4;   // 51.2 MB (pooled/t1/t2)
    __bf16* bfbuf = (__bf16*)w; w += (size_t)N_NODES * FDIM * 2;   // 25.6 MB
    __bf16* wp    = (__bf16*)w; w += (size_t)8 * 16384 * 2;        // packed weights
    float*  stats = (float*)w;  w += (size_t)2048 * 4;             // 8 BN stat sets

    const int EL = N_NODES * FDIM / 256;   // 50000 elementwise blocks (exact)

    // stats must be zero before atomics; re-zeroed every launch (graph-safe)
    k_zero<<<8, 256, 0, stream>>>(stats, 2048);

    Ptr8 mats;
    for (int l = 0; l < NLAYER; ++l) {
        mats.p[2 * l]     = (const float*)d_in[2 + 8 * l];      // W1
        mats.p[2 * l + 1] = (const float*)d_in[2 + 8 * l + 1];  // W2
    }
    k_repack<<<256, 32, 0, stream>>>(wp, mats);

    Ptr5 bs;
    for (int l = 0; l < 5; ++l) bs.p[l] = (const float*)d_in[35 + 2 * l];
    k_init_score<<<4, 256, 0, stream>>>((float*)d_out, bs);

    // readout contribution of hidden_rep[0] = x
    k_pool_pred<<<NGRAPH, 256, 0, stream>>>(x, (const float*)d_in[34], (float*)d_out);

    for (int l = 0; l < NLAYER; ++l) {
        int base = 2 + 8 * l;
        const float* b1    = (const float*)d_in[base + 2];
        const float* b2    = (const float*)d_in[base + 3];
        const float* bn1_b = (const float*)d_in[base + 4];
        const float* bn1_g = (const float*)d_in[base + 5];
        const float* bn_b  = (const float*)d_in[base + 6];
        const float* bn_g  = (const float*)d_in[base + 7];
        const float* hin   = (l == 0) ? x : hbuf;

        // aggregation: pooled = (1+eps)h + sum_{e:row=i} h[col]
        k_scale_init<<<EL, 256, 0, stream>>>(tbuf, hin, eps, l);
        k_edge_scatter<<<N_EDGES * 32 / 256, 256, 0, stream>>>(tbuf, hin, erow, ecol);

        // MLP layer 1: bf16 WMMA GEMM + bias, then BN stats + BN/ReLU -> bf16
        k_tobf16<<<EL, 256, 0, stream>>>(bfbuf, tbuf);
        k_gemm_bias<<<N_NODES / 16, 256, 0, stream>>>(
            tbuf, bfbuf, wp + (size_t)(2 * l) * 16384, b1);
        float* st1 = stats + (size_t)(2 * l) * 256;
        k_colstats<<<N_NODES / 100, 256, 0, stream>>>(st1, tbuf);
        k_bn_relu_bf16<<<EL, 256, 0, stream>>>(bfbuf, tbuf, st1, bn1_g, bn1_b);

        // MLP layer 2 + outer BN/ReLU -> f32 h
        k_gemm_bias<<<N_NODES / 16, 256, 0, stream>>>(
            tbuf, bfbuf, wp + (size_t)(2 * l + 1) * 16384, b2);
        float* st2 = stats + (size_t)(2 * l + 1) * 256;
        k_colstats<<<N_NODES / 100, 256, 0, stream>>>(st2, tbuf);
        k_bn_relu_f32<<<EL, 256, 0, stream>>>(hbuf, tbuf, st2, bn_g, bn_b);

        // readout contribution of hidden_rep[l+1]
        k_pool_pred<<<NGRAPH, 256, 0, stream>>>(
            hbuf, (const float*)d_in[34 + 2 * (l + 1)], (float*)d_out);
    }
}
